// Head_2903397892222
// MI455X (gfx1250) — matrix operand order repaired and checked by reading, not compiled, not run on hardware
//
#include <hip/hip_runtime.h>

typedef _Float16 f16;
typedef __attribute__((ext_vector_type(16))) _Float16 v16h;
typedef __attribute__((ext_vector_type(8)))  _Float16 v8h;
typedef __attribute__((ext_vector_type(8)))  float    v8f;
typedef __attribute__((ext_vector_type(4)))  float    v4f;

#define WMMA_F16(a, b, c) \
  __builtin_amdgcn_wmma_f32_16x16x32_f16(false, (a), false, (b), (short)0, (c), false, false)

static constexpr int kB  = 8;
static constexpr int kT  = 2048;
static constexpr int kD  = 1024;
static constexpr int kHS = 64;
static constexpr float kLog2e = 1.44269504088896340736f;

// Packed-W geometry: tile = (kcIdx, w, nb) -> 32x16 f16 B-layout tile, 512 f16.
static constexpr int kKC     = kD / 32;        // 32 K-chunks
static constexpr int kWTiles = kKC * 3 * 4;    // 384 tiles, 1 KB each

// ---------------------------------------------------------------------------
// Kernel 0: pack Wq/Wk/Wv (fp32 [D][HS]) into f16 WMMA-B-layout tiles, once.
// B tile order used here (32x16, contiguous k): lane&15 = column, halves i=0..15 = K half*16 + i.
// ---------------------------------------------------------------------------
__global__ __launch_bounds__(128, 1)
void pack_w_kernel(const float* __restrict__ Wq,
                   const float* __restrict__ Wk,
                   const float* __restrict__ Wv,
                   f16* __restrict__ pw)
{
  const int lane = threadIdx.x & 31;
  const int wv   = threadIdx.x >> 5;
  const int tile = blockIdx.x * 4 + wv;        // 0 .. kWTiles-1
  if (tile >= kWTiles) return;
  const int kcIdx = tile / 12;
  const int rem   = tile % 12;
  const int w     = rem >> 2;                  // 0=q 1=k 2=v
  const int nb    = rem & 3;
  const int half  = lane >> 4;
  const int l15   = lane & 15;

  const float* W = (w == 0) ? Wq : (w == 1) ? Wk : Wv;
  const float* src = W + (size_t)(kcIdx * 32 + half * 16) * kHS + nb * 16 + l15;

  v16h b;
#pragma unroll
  for (int i = 0; i < 16; ++i) b[i] = (f16)src[(size_t)i * kHS];
  *(v16h*)(pw + (size_t)tile * 512 + lane * 16) = b;
}

// ---------------------------------------------------------------------------
// Kernel 1: fused QKV projection.  One wave -> 16 rows x (3 x 64) outputs.
// x read once; W tiles pre-packed in B layout; q gets 1/32 scale folded in;
// V stored transposed ([B][HS][T] f16) for contiguous PV B-tiles.
// ---------------------------------------------------------------------------
__global__ __launch_bounds__(128, 1)
void qkv_proj_kernel(const float* __restrict__ x,
                     const f16* __restrict__ pw,
                     f16* __restrict__ qo, f16* __restrict__ ko, f16* __restrict__ vt)
{
  const int lane = threadIdx.x & 31;
  const int wv   = threadIdx.x >> 5;
  const int tile = blockIdx.x * 4 + wv;        // 0 .. B*T/16-1
  const int row0 = tile * 16;
  const int half = lane >> 4;
  const int l15  = lane & 15;

  v8f accq[4] = {v8f{}, v8f{}, v8f{}, v8f{}};
  v8f acck[4] = {v8f{}, v8f{}, v8f{}, v8f{}};
  v8f accv[4] = {v8f{}, v8f{}, v8f{}, v8f{}};

  const float* xr = x + (size_t)(row0 + l15) * kD;

  for (int kc = 0; kc < kD; kc += 32) {
    v4f x0 = *(const v4f*)(xr + kc + half * 16);
    v4f x1 = *(const v4f*)(xr + kc + half * 16 + 4);
    v4f x2 = *(const v4f*)(xr + kc + half * 16 + 8);
    v4f x3 = *(const v4f*)(xr + kc + half * 16 + 12);
    v16h a;
#pragma unroll
    for (int i = 0; i < 4; ++i) {
      a[i]      = (f16)x0[i];
      a[4 + i]  = (f16)x1[i];
      a[8 + i]  = (f16)x2[i];
      a[12 + i] = (f16)x3[i];
    }
    const f16* base = pw + (size_t)((kc >> 5) * 12) * 512 + lane * 16;
#pragma unroll
    for (int nb = 0; nb < 4; ++nb) {
      v16h bq = *(const v16h*)(base + (size_t)(0 * 4 + nb) * 512);
      v16h bk = *(const v16h*)(base + (size_t)(1 * 4 + nb) * 512);
      v16h bv = *(const v16h*)(base + (size_t)(2 * 4 + nb) * 512);
      accq[nb] = WMMA_F16(a, bq, accq[nb]);
      acck[nb] = WMMA_F16(a, bk, acck[nb]);
      accv[nb] = WMMA_F16(a, bv, accv[nb]);
    }
  }

  const float scale = 0.03125f;                // D^-0.5 = 1/32 exactly
  const int  b    = row0 >> 11;                // 2048 rows per batch
  const int  trow = row0 & (kT - 1);
#pragma unroll
  for (int nb = 0; nb < 4; ++nb) {
#pragma unroll
    for (int r = 0; r < 8; ++r) {
      const size_t idx = (size_t)(row0 + r + half * 8) * kHS + nb * 16 + l15;
      qo[idx] = (f16)(accq[nb][r] * scale);
      ko[idx] = (f16)(acck[nb][r]);
      vt[((size_t)b * kHS + nb * 16 + l15) * kT + trow + r + half * 8] = (f16)(accv[nb][r]);
    }
  }
}

// ---------------------------------------------------------------------------
// Flash-attention key-block body: 64 keys, 8 score WMMAs + 8 PV WMMAs.
// MASK=true only for the diagonal blocks of the owning q-tile.
// ---------------------------------------------------------------------------
template <bool MASK>
__device__ __forceinline__
void attn_block(int kbase, int q0, int half, int l15,
                const v16h& aq0, const v16h& aq1,
                const f16* __restrict__ Kb, const f16* __restrict__ VtB,
                f16* __restrict__ pl,
                float (&m8)[8], float (&l8)[8], v8f (&o)[4])
{
  // ---- scores: four 16x16 tiles, each K=64 via two chained WMMAs
  v8f s[4];
#pragma unroll
  for (int t = 0; t < 4; ++t) {
    const int key = kbase + t * 16 + l15;      // B layout: lane&15 = column (key)
    const f16* kr = Kb + (size_t)key * kHS;
    v16h b0 = *(const v16h*)(kr + half * 16);
    v16h b1 = *(const v16h*)(kr + 32 + half * 16);
    v8f z = {};
    v8f ss = WMMA_F16(aq0, b0, z);
    ss = WMMA_F16(aq1, b1, ss);
    if constexpr (MASK) {
#pragma unroll
      for (int r = 0; r < 8; ++r) {            // causal mask in C/D layout
        const int qrow = q0 + r + half * 8;
        if (key > qrow) ss[r] = -INFINITY;
      }
    }
    s[t] = ss;
  }

  // ---- online softmax (per-row, half-wave width-16 reductions, wave32)
  float corr[8];
#pragma unroll
  for (int r = 0; r < 8; ++r) {
    float rm = fmaxf(fmaxf(s[0][r], s[1][r]), fmaxf(s[2][r], s[3][r]));
#pragma unroll
    for (int off = 1; off < 16; off <<= 1)
      rm = fmaxf(rm, __shfl_xor(rm, off, 16));
    const float mn = fmaxf(m8[r], rm);
    corr[r] = exp2f((m8[r] - mn) * kLog2e);    // 0 on first block (m8 = -inf)
    m8[r] = mn;
    float rs = 0.0f;
#pragma unroll
    for (int t = 0; t < 4; ++t) {
      const float p = exp2f((s[t][r] - mn) * kLog2e);
      s[t][r] = p;
      rs += p;
    }
#pragma unroll
    for (int off = 1; off < 16; off <<= 1)
      rs += __shfl_xor(rs, off, 16);
    l8[r] = l8[r] * corr[r] + rs;
  }
#pragma unroll
  for (int blk = 0; blk < 4; ++blk)
#pragma unroll
    for (int r = 0; r < 8; ++r) o[blk][r] *= corr[r];

  // ---- transpose P (16x64): C/D layout -> A layout via per-wave LDS tile
#pragma unroll
  for (int t = 0; t < 4; ++t)
#pragma unroll
    for (int r = 0; r < 8; ++r)
      pl[(r + half * 8) * 64 + t * 16 + l15] = (f16)s[t][r];
  asm volatile("s_wait_dscnt 0" ::: "memory");
  v16h ap0, ap1;
  {
    v8h c0 = *(const v8h*)(pl + l15 * 64 + half * 16);
    v8h c1 = *(const v8h*)(pl + l15 * 64 + 8 + half * 16);
    v8h c2 = *(const v8h*)(pl + l15 * 64 + 32 + half * 16);
    v8h c3 = *(const v8h*)(pl + l15 * 64 + 40 + half * 16);
#pragma unroll
    for (int i = 0; i < 8; ++i) {
      ap0[i] = c0[i]; ap0[8 + i] = c1[i];
      ap1[i] = c2[i]; ap1[8 + i] = c3[i];
    }
  }

  // ---- PV: 16x64 @ 64x64; V^T gives contiguous B-tiles (32 B / lane)
#pragma unroll
  for (int blk = 0; blk < 4; ++blk) {
    const f16* vp = VtB + (size_t)(blk * 16 + l15) * kT + kbase + half * 16;
    v16h bv0 = *(const v16h*)(vp);
    v16h bv1 = *(const v16h*)(vp + 32);
    o[blk] = WMMA_F16(ap0, bv0, o[blk]);
    o[blk] = WMMA_F16(ap1, bv1, o[blk]);
  }
}

// ---------------------------------------------------------------------------
// Kernel 2: flash attention.  One BLOCK (4 waves) per 16-query tile; keys are
// split round-robin across the 4 waves (4096 waves total -> real latency
// hiding + balanced causal tail).  Wave-private (m,l,O) partials are merged
// with a log-sum-exp combine through LDS after a barrier.
// Blocks are mapped long-tiles-first to shrink the scheduling tail.
// ---------------------------------------------------------------------------
__global__ __launch_bounds__(128, 1)
void attn_kernel(const f16* __restrict__ Q,
                 const f16* __restrict__ K,
                 const f16* __restrict__ Vt,
                 float* __restrict__ out)
{
  __shared__ f16   plds[4][16 * 64];           // per-wave P transpose tiles (8 KB)
  __shared__ float pm [4][8][32];              // partial row-max   (4 KB)
  __shared__ float pls[4][8][32];              // partial row-sum   (4 KB)
  __shared__ float po [4][4][8][32];           // partial O         (16 KB)

  const int lane = threadIdx.x & 31;
  const int wv   = threadIdx.x >> 5;
  const int b    = blockIdx.x >> 7;            // 128 q-tiles per batch
  const int q0   = (127 - (blockIdx.x & 127)) * 16;  // long tiles first
  const int half = lane >> 4;
  const int l15  = lane & 15;

  const f16* Qb  = Q  + (size_t)b * kT * kHS;
  const f16* Kb  = K  + (size_t)b * kT * kHS;
  const f16* VtB = Vt + (size_t)b * kHS * kT;
  f16* pl = &plds[wv][0];

  // ---- load q tile (16x64) into two A-layout chunks (kdim 0..31, 32..63)
  v16h aq0, aq1;
  {
    const f16* qr = Qb + (size_t)(q0 + l15) * kHS;
    v8h t0 = *(const v8h*)(qr + half * 16);
    v8h t1 = *(const v8h*)(qr + 8 + half * 16);
    v8h t2 = *(const v8h*)(qr + 32 + half * 16);
    v8h t3 = *(const v8h*)(qr + 40 + half * 16);
#pragma unroll
    for (int i = 0; i < 8; ++i) {
      aq0[i] = t0[i]; aq0[8 + i] = t1[i];
      aq1[i] = t2[i]; aq1[8 + i] = t3[i];
    }
  }

  float m8[8], l8[8];
  v8f o[4] = {v8f{}, v8f{}, v8f{}, v8f{}};
#pragma unroll
  for (int r = 0; r < 8; ++r) { m8[r] = -INFINITY; l8[r] = 0.0f; }

  const int nblk  = (q0 + 79) >> 6;            // blocks needed (keys <= q0+15)
  const int nfull = (q0 + 1) >> 6;             // blocks fully below the diagonal

  // keys split round-robin over the 4 waves of this block
  for (int jb = wv; jb < nblk; jb += 4) {
    const int knext = (jb + 4) * 64;
    if (knext < nblk * 64) {                   // prefetch this wave's next block
      __builtin_prefetch(Kb + (size_t)(knext + l15) * kHS, 0, 3);
      __builtin_prefetch(VtB + (size_t)l15 * kT + knext, 0, 3);
    }
    if (jb < nfull)
      attn_block<false>(jb * 64, q0, half, l15, aq0, aq1, Kb, VtB, pl, m8, l8, o);
    else
      attn_block<true >(jb * 64, q0, half, l15, aq0, aq1, Kb, VtB, pl, m8, l8, o);
  }

  // ---- publish partial state, merge with log-sum-exp combine
#pragma unroll
  for (int r = 0; r < 8; ++r) {
    pm [wv][r][lane] = m8[r];
    pls[wv][r][lane] = l8[r];
#pragma unroll
    for (int blk = 0; blk < 4; ++blk)
      po[wv][blk][r][lane] = o[blk][r];
  }
  __syncthreads();

  if (wv == 0) {
#pragma unroll
    for (int r = 0; r < 8; ++r) {
      float M = pm[0][r][lane];
#pragma unroll
      for (int w = 1; w < 4; ++w) M = fmaxf(M, pm[w][r][lane]);
      float e[4], L = 0.0f;
#pragma unroll
      for (int w = 0; w < 4; ++w) {
        e[w] = exp2f((pm[w][r][lane] - M) * kLog2e);   // 0 for empty waves
        L += pls[w][r][lane] * e[w];
      }
      const float invL = 1.0f / L;
#pragma unroll
      for (int blk = 0; blk < 4; ++blk) {
        float O = 0.0f;
#pragma unroll
        for (int w = 0; w < 4; ++w) O += po[w][blk][r][lane] * e[w];
        out[((size_t)b * kT + q0 + r + half * 8) * kHS + blk * 16 + l15] = O * invL;
      }
    }
  }
}

// ---------------------------------------------------------------------------
extern "C" void kernel_launch(void* const* d_in, const int* in_sizes, int n_in,
                              void* d_out, int out_size, void* d_ws, size_t ws_size,
                              hipStream_t stream)
{
  (void)in_sizes; (void)n_in; (void)out_size; (void)ws_size;
  const float* x  = (const float*)d_in[0];
  const float* Wq = (const float*)d_in[1];
  const float* Wk = (const float*)d_in[2];
  const float* Wv = (const float*)d_in[3];
  float* out = (float*)d_out;

  const size_t nQKV = (size_t)kB * kT * kHS;   // 1,048,576 elements each
  f16* q  = (f16*)d_ws;
  f16* k  = q  + nQKV;
  f16* vt = k  + nQKV;                         // transposed V
  f16* pw = vt + nQKV;                         // packed W tiles (384 KB)

  pack_w_kernel<<<(kWTiles + 3) / 4, 128, 0, stream>>>(Wq, Wk, Wv, pw);

  const int tiles = kB * kT / 16;              // 1024 q-tiles
  qkv_proj_kernel<<<tiles / 4, 128, 0, stream>>>(x, pw, q, k, vt);
  attn_kernel<<<tiles, 128, 0, stream>>>(q, k, vt, out);
}
